// Attention_37589553775245
// MI455X (gfx1250) — matrix operand order repaired and checked by reading, not compiled, not run on hardware
//
#include <hip/hip_runtime.h>

// Sizes from the reference
#define B_    8
#define N_    1024
#define DIM_  512
#define H_    8
#define DH_   64
#define HD_   512      // H*DH
#define QKV3_ 1536     // 3*H*DH

typedef __attribute__((ext_vector_type(16))) _Float16 v16h;
typedef __attribute__((ext_vector_type(8)))  float    v8f;

union frag_u { uint4 u[2]; v16h h; };

__device__ __forceinline__ v8f wmma16(v16h a, v16h b, v8f c) {
  // D = A(16x32 f16) * B(32x16 f16) + C(16x16 f32)
  return __builtin_amdgcn_wmma_f32_16x16x32_f16(false, a, false, b, (short)0, c, false, false);
}

// ---- fragment loaders (wave32 WMMA; A and B share the contiguous k order k = 16*half + e) ----
// All loaders read two contiguous 16B runs per lane -> global_load_b128 / ds_load_b128.

// A fragment 16x32 f16 from row-major f16 [*, ld] (ld mult of 32), origin (row0, k0 mult 32).
__device__ __forceinline__ v16h load_a_frag(const _Float16* __restrict__ p, int ld, int row0, int k0) {
  int l = threadIdx.x & 31;
  const _Float16* rp = p + (size_t)(row0 + (l & 15)) * ld + k0 + (l >> 4) * 16;
  frag_u f;
  f.u[0] = *(const uint4*)(rp);        // elements 0..7
  f.u[1] = *(const uint4*)(rp + 8);    // elements 8..15
  return f.h;
}

// B fragment 32x16 where source is row-major [N, K] (transpose view, contiguous in K).
__device__ __forceinline__ v16h load_bT_frag(const _Float16* __restrict__ p, int ld, int k0, int col0) {
  int l = threadIdx.x & 31;
  const _Float16* rp = p + (size_t)(col0 + (l & 15)) * ld + k0 + (l >> 4) * 16;
  frag_u f;
  f.u[0] = *(const uint4*)(rp);
  f.u[1] = *(const uint4*)(rp + 8);
  return f.h;
}

// B fragment from pre-swizzled (fragment-ordered) weight storage:
// tile (kt, nt) stored as [32 lanes][16 f16] contiguous.
__device__ __forceinline__ v16h load_b_swz(const _Float16* __restrict__ swz, int ktiles, int kt, int nt) {
  int l = threadIdx.x & 31;
  const _Float16* p = swz + (((size_t)nt * ktiles + kt) * 32 + l) * 16;
  frag_u f;
  f.u[0] = *(const uint4*)(p);
  f.u[1] = *(const uint4*)(p + 8);
  return f.h;
}

// ---------------- prep: f32 -> f16 convert ----------------
__global__ void cvt_f32_f16_kernel(const float* __restrict__ src, _Float16* __restrict__ dst, int n) {
  int i = blockIdx.x * blockDim.x + threadIdx.x;
  if (i < n) dst[i] = (_Float16)src[i];
}

// ---------------- prep: swizzle weight [K,Ncols] f32 into fragment-ordered f16 ----------------
__global__ void swizzle_w_kernel(const float* __restrict__ w, _Float16* __restrict__ swz,
                                 int K, int Ncols) {
  int idx = blockIdx.x * blockDim.x + threadIdx.x;
  if (idx >= K * Ncols) return;
  int k = idx / Ncols, n = idx % Ncols;
  int kt = k >> 5, nt = n >> 4;
  int lane = (n & 15) + ((k >> 4) & 1) * 16;   // same k order as A: lanes 16-31 hold K 16-31 of the tile
  int e = k & 15;
  swz[(((size_t)nt * (K >> 5) + kt) * 32 + lane) * 16 + e] = (_Float16)w[idx];
}

// ---------------- GCN degree normalization: dinv[i] = (nnz(row_i)+1)^-1/2 ----------------
__global__ void dinv_kernel(const float* __restrict__ adj, float* __restrict__ dinv) {
  int row = (blockIdx.x * blockDim.x + threadIdx.x) >> 5;   // one wave32 per row
  int l = threadIdx.x & 31;
  if (row >= N_) return;
  const float* rp = adj + (size_t)row * N_;
  float cnt = 0.f;
  for (int j = l; j < N_; j += 32) cnt += (rp[j] != 0.f) ? 1.f : 0.f;  // ceil(adj) binarizes
#pragma unroll
  for (int off = 16; off > 0; off >>= 1) cnt += __shfl_down(cnt, off, 32);
  if (l == 0) dinv[row] = rsqrtf(cnt + 1.0f);               // +1 for identity
}

// ---------------- QKV GEMM: [8192,512] @ [512,1536], 128x128 block, 2x4 wave tiles -----------
// q,k stored head-major [bh][n][dh]; v stored TRANSPOSED head-major [bh][dh][n] so the
// attention attn@V B-fragments are lane-contiguous.
__global__ void qkv_gemm_kernel(const _Float16* __restrict__ xf, const _Float16* __restrict__ wswz,
                                _Float16* __restrict__ qb, _Float16* __restrict__ kb,
                                _Float16* __restrict__ vb) {
  const int wv = threadIdx.x >> 5;
  const int tr = wv & 3;                       // 4 row groups of 32
  const int tcg = wv >> 2;                     // 2 col groups of 64
  const int m_base = blockIdx.x * 128 + tr * 32;
  const int n_base = blockIdx.y * 128 + tcg * 64;
  const int l  = threadIdx.x & 31;
  const int cn = l & 15, mh = l >> 4;
  const int ktiles = DIM_ >> 5;
  const int nt0 = n_base >> 4;

  v8f acc[2][4];
#pragma unroll
  for (int i = 0; i < 2; ++i)
#pragma unroll
    for (int j = 0; j < 4; ++j) acc[i][j] = (v8f){};

#pragma unroll 2
  for (int kk = 0; kk < DIM_; kk += 32) {
    int kt = kk >> 5;
    v16h a0 = load_a_frag(xf, DIM_, m_base,      kk);
    v16h a1 = load_a_frag(xf, DIM_, m_base + 16, kk);
    v16h b0 = load_b_swz(wswz, ktiles, kt, nt0);
    v16h b1 = load_b_swz(wswz, ktiles, kt, nt0 + 1);
    v16h b2 = load_b_swz(wswz, ktiles, kt, nt0 + 2);
    v16h b3 = load_b_swz(wswz, ktiles, kt, nt0 + 3);
    acc[0][0] = wmma16(a0, b0, acc[0][0]);
    acc[0][1] = wmma16(a0, b1, acc[0][1]);
    acc[0][2] = wmma16(a0, b2, acc[0][2]);
    acc[0][3] = wmma16(a0, b3, acc[0][3]);
    acc[1][0] = wmma16(a1, b0, acc[1][0]);
    acc[1][1] = wmma16(a1, b1, acc[1][1]);
    acc[1][2] = wmma16(a1, b2, acc[1][2]);
    acc[1][3] = wmma16(a1, b3, acc[1][3]);
  }

  // col group [n_base, n_base+64) is 64-aligned: which & head uniform per wave
  const int which = (n_base >> 9);             // 0=q,1=k,2=v
  const int hh = (n_base >> 6) & 7;
#pragma unroll
  for (int j = 0; j < 4; ++j) {
    int dd = j * 16 + cn;                      // dh index within the head
#pragma unroll
    for (int i = 0; i < 2; ++i) {
      v8f cc = acc[i][j];
      int row0 = m_base + i * 16 + 8 * mh;     // C layout: VGPR r -> M = r + 8*half
      int bb = row0 >> 10, nn0 = row0 & 1023;
      if (which == 2) {                        // V: transposed layout -> packed 16B store
        union { _Float16 hh8[8]; uint4 u; } pk;
#pragma unroll
        for (int r = 0; r < 8; ++r) pk.hh8[r] = (_Float16)cc[r];
        *(uint4*)(vb + (((size_t)bb * H_ + hh) * DH_ + dd) * N_ + nn0) = pk.u;
      } else {
        _Float16* dst = (which == 0) ? qb : kb;
#pragma unroll
        for (int r = 0; r < 8; ++r)
          dst[(((size_t)bb * H_ + hh) * N_ + (nn0 + r)) * DH_ + dd] = (_Float16)cc[r];
      }
    }
  }
}

// ---------------- attention: full 64x1024 fp32 score panel in LDS (256KB of 320KB WGP LDS) ----
__global__ void attn_kernel(const _Float16* __restrict__ qb, const _Float16* __restrict__ kb,
                            const _Float16* __restrict__ vt, const float* __restrict__ adj,
                            const float* __restrict__ dinv, _Float16* __restrict__ aout) {
  extern __shared__ char smem[];
  float* S   = (float*)smem;                       // [64][1024] scores
  float* red = (float*)(smem + 64 * 1024 * 4);     // [64][8] softmax scratch

  const int bid  = blockIdx.x;
  const int rt   = bid & 15;          // 16 row-tiles of 64 per (b,h)
  const int bh   = bid >> 4;
  const int b    = bh >> 3, h = bh & 7;
  const int row0 = rt * 64;
  const _Float16* Q  = qb + (size_t)bh * N_ * DH_;   // [n][dh]
  const _Float16* K  = kb + (size_t)bh * N_ * DH_;   // [n][dh]
  const _Float16* Vt = vt + (size_t)bh * DH_ * N_;   // [dh][n]  (transposed)

  const int wv = threadIdx.x >> 5;
  const int rs = wv & 3;              // row subtile 0..3
  const int cg = wv >> 2;             // column-group 0/1
  const int l  = threadIdx.x & 31;
  const int cn = l & 15, mh = l >> 4;
  const float scale = 0.125f;         // DH^-0.5

  // --- phase A: S = scale * Q K^T via WMMA (K-dim 64 = 2 chained wmma) ---
  v16h qa0 = load_a_frag(Q, DH_, row0 + rs * 16, 0);
  v16h qa1 = load_a_frag(Q, DH_, row0 + rs * 16, 32);
#pragma unroll 2
  for (int ct = cg; ct < 64; ct += 2) {
    v16h kb0 = load_bT_frag(K, DH_, 0,  ct * 16);
    v16h kb1 = load_bT_frag(K, DH_, 32, ct * 16);
    v8f c = {};
    c = wmma16(qa0, kb0, c);
    c = wmma16(qa1, kb1, c);
#pragma unroll
    for (int r = 0; r < 8; ++r) {
      int m = rs * 16 + r + 8 * mh;
      S[m * 1024 + ct * 16 + cn] = c[r] * scale;
    }
  }
  __syncthreads();

  // --- phase B: softmax over 1024 cols (4 threads/row, float4-vectorized) + GCN bias ---
  {
    int r   = threadIdx.x >> 2;
    int seg = threadIdx.x & 3;
    float4* s4 = (float4*)(S + r * 1024 + seg * 256);
    float mx = -1e30f;
    for (int jj = 0; jj < 64; ++jj) {
      float4 v = s4[jj];
      mx = fmaxf(mx, fmaxf(fmaxf(v.x, v.y), fmaxf(v.z, v.w)));
    }
    red[r * 8 + seg] = mx;
    __syncthreads();
    mx = fmaxf(fmaxf(red[r * 8 + 0], red[r * 8 + 1]), fmaxf(red[r * 8 + 2], red[r * 8 + 3]));
    float sum = 0.f;
    for (int jj = 0; jj < 64; ++jj) {
      float4 v = s4[jj];
      v.x = __expf(v.x - mx); v.y = __expf(v.y - mx);
      v.z = __expf(v.z - mx); v.w = __expf(v.w - mx);
      sum += (v.x + v.y) + (v.z + v.w);
      s4[jj] = v;
    }
    red[r * 8 + 4 + seg] = sum;
    __syncthreads();
    sum = red[r * 8 + 4] + red[r * 8 + 5] + red[r * 8 + 6] + red[r * 8 + 7];
    float inv = 1.f / sum;
    int gi = row0 + r;
    float di = dinv[gi];
    const float4* a4 = (const float4*)(adj + (size_t)gi * N_ + seg * 256);
    const float4* d4 = (const float4*)(dinv + seg * 256);
    for (int jj = 0; jj < 64; ++jj) {
      float4 v = s4[jj];
      float4 av = a4[jj];
      float4 dv = d4[jj];
      int j = seg * 256 + jj * 4;
      v.x = v.x * inv + di * dv.x * ((av.x != 0.f ? 1.f : 0.f) + (j + 0 == gi ? 1.f : 0.f));
      v.y = v.y * inv + di * dv.y * ((av.y != 0.f ? 1.f : 0.f) + (j + 1 == gi ? 1.f : 0.f));
      v.z = v.z * inv + di * dv.z * ((av.z != 0.f ? 1.f : 0.f) + (j + 2 == gi ? 1.f : 0.f));
      v.w = v.w * inv + di * dv.w * ((av.w != 0.f ? 1.f : 0.f) + (j + 3 == gi ? 1.f : 0.f));
      s4[jj] = v;
    }
  }
  __syncthreads();

  // --- phase C: out = attn @ V via WMMA; A from LDS (ds_load_b128 pairs + cvt), B from Vt ---
  const int cs0 = cg * 2;             // this wave's first dh subtile (0 or 2)
  v8f o0 = {}, o1 = {};
#pragma unroll 2
  for (int kk = 0; kk < N_; kk += 32) {
    v16h a;
    {
      const float* sp = S + (rs * 16 + cn) * 1024 + kk + mh * 16;
      float4 f0 = *(const float4*)(sp);        // k  +0..3
      float4 f1 = *(const float4*)(sp + 4);    // k  +4..7
      float4 f2 = *(const float4*)(sp + 8);    // k  +8..11
      float4 f3 = *(const float4*)(sp + 12);   // k  +12..15
      a[0]  = (_Float16)f0.x; a[1]  = (_Float16)f0.y; a[2]  = (_Float16)f0.z; a[3]  = (_Float16)f0.w;
      a[4]  = (_Float16)f1.x; a[5]  = (_Float16)f1.y; a[6]  = (_Float16)f1.z; a[7]  = (_Float16)f1.w;
      a[8]  = (_Float16)f2.x; a[9]  = (_Float16)f2.y; a[10] = (_Float16)f2.z; a[11] = (_Float16)f2.w;
      a[12] = (_Float16)f3.x; a[13] = (_Float16)f3.y; a[14] = (_Float16)f3.z; a[15] = (_Float16)f3.w;
    }
    v16h vb0 = load_bT_frag(Vt, N_, kk, cs0 * 16);        // contiguous: Vt[d][k..k+15]
    v16h vb1 = load_bT_frag(Vt, N_, kk, cs0 * 16 + 16);
    o0 = wmma16(a, vb0, o0);
    o1 = wmma16(a, vb1, o1);
  }
#pragma unroll
  for (int r = 0; r < 8; ++r) {
    int m  = row0 + rs * 16 + r + 8 * mh;
    int d0 = h * DH_ + cs0 * 16 + cn;
    aout[((size_t)b * N_ + m) * HD_ + d0]      = (_Float16)o0[r];
    aout[((size_t)b * N_ + m) * HD_ + d0 + 16] = (_Float16)o1[r];
  }
}

// ---------------- output projection: [8192,512] @ [512,512] + bias, fp32 out ----------------
__global__ void proj_gemm_kernel(const _Float16* __restrict__ af, const _Float16* __restrict__ wswz,
                                 const float* __restrict__ bias, float* __restrict__ out) {
  const int wv = threadIdx.x >> 5;
  const int tr = wv & 3;
  const int tcg = wv >> 2;
  const int m_base = blockIdx.x * 128 + tr * 32;
  const int n_base = blockIdx.y * 128 + tcg * 64;
  const int l  = threadIdx.x & 31;
  const int cn = l & 15, mh = l >> 4;
  const int ktiles = HD_ >> 5;
  const int nt0 = n_base >> 4;

  v8f acc[2][4];
#pragma unroll
  for (int i = 0; i < 2; ++i)
#pragma unroll
    for (int j = 0; j < 4; ++j) acc[i][j] = (v8f){};

#pragma unroll 2
  for (int kk = 0; kk < HD_; kk += 32) {
    int kt = kk >> 5;
    v16h a0 = load_a_frag(af, HD_, m_base,      kk);
    v16h a1 = load_a_frag(af, HD_, m_base + 16, kk);
    v16h b0 = load_b_swz(wswz, ktiles, kt, nt0);
    v16h b1 = load_b_swz(wswz, ktiles, kt, nt0 + 1);
    v16h b2 = load_b_swz(wswz, ktiles, kt, nt0 + 2);
    v16h b3 = load_b_swz(wswz, ktiles, kt, nt0 + 3);
    acc[0][0] = wmma16(a0, b0, acc[0][0]);
    acc[0][1] = wmma16(a0, b1, acc[0][1]);
    acc[0][2] = wmma16(a0, b2, acc[0][2]);
    acc[0][3] = wmma16(a0, b3, acc[0][3]);
    acc[1][0] = wmma16(a1, b0, acc[1][0]);
    acc[1][1] = wmma16(a1, b1, acc[1][1]);
    acc[1][2] = wmma16(a1, b2, acc[1][2]);
    acc[1][3] = wmma16(a1, b3, acc[1][3]);
  }
#pragma unroll
  for (int j = 0; j < 4; ++j) {
    int col = n_base + j * 16 + cn;
    float bv = bias[col];
#pragma unroll
    for (int i = 0; i < 2; ++i) {
      v8f cc = acc[i][j];
#pragma unroll
      for (int r = 0; r < 8; ++r) {
        int rowg = m_base + i * 16 + r + 8 * mh;
        out[(size_t)rowg * DIM_ + col] = cc[r] + bv;
      }
    }
  }
}

extern "C" void kernel_launch(void* const* d_in, const int* in_sizes, int n_in,
                              void* d_out, int out_size, void* d_ws, size_t ws_size,
                              hipStream_t stream) {
  const float* x     = (const float*)d_in[0];   // [8,1024,512]
  const float* adj   = (const float*)d_in[1];   // [1024,1024]
  const float* w_qkv = (const float*)d_in[2];   // [512,1536]
  const float* w_out = (const float*)d_in[3];   // [512,512]
  const float* b_out = (const float*)d_in[4];   // [512]
  float* out = (float*)d_out;                   // [8,1024,512] fp32

  char* ws = (char*)d_ws;
  size_t off = 0;
  auto carve = [&](size_t bytes) { void* p = ws + off; off += (bytes + 255) & ~(size_t)255; return p; };
  _Float16* xf    = (_Float16*)carve((size_t)B_ * N_ * DIM_ * 2);      // 8 MB
  _Float16* wqkvs = (_Float16*)carve((size_t)DIM_ * QKV3_ * 2);        // 1.5 MB (swizzled)
  _Float16* wouts = (_Float16*)carve((size_t)HD_ * DIM_ * 2);          // 0.5 MB (swizzled)
  _Float16* qf    = (_Float16*)carve((size_t)B_ * H_ * N_ * DH_ * 2);  // 8 MB
  _Float16* kf    = (_Float16*)carve((size_t)B_ * H_ * N_ * DH_ * 2);
  _Float16* vtf   = (_Float16*)carve((size_t)B_ * H_ * N_ * DH_ * 2);  // transposed [bh][dh][n]
  _Float16* aof   = (_Float16*)carve((size_t)B_ * N_ * HD_ * 2);       // 8 MB
  float*    dnv   = (float*)carve((size_t)N_ * 4);
  (void)ws_size; (void)in_sizes; (void)n_in; (void)out_size;           // ~42 MB total

  // 1) prep: x -> f16, weights -> fragment-ordered f16 (B-fragments become 2x b128 loads)
  {
    int n1 = B_ * N_ * DIM_;
    cvt_f32_f16_kernel<<<(n1 + 255) / 256, 256, 0, stream>>>(x, xf, n1);
    int n2 = DIM_ * QKV3_;
    swizzle_w_kernel<<<(n2 + 255) / 256, 256, 0, stream>>>(w_qkv, wqkvs, DIM_, QKV3_);
    int n3 = HD_ * DIM_;
    swizzle_w_kernel<<<(n3 + 255) / 256, 256, 0, stream>>>(w_out, wouts, HD_, DIM_);
  }
  // 2) degree normalization vector
  dinv_kernel<<<N_ / 8, 256, 0, stream>>>(adj, dnv);
  // 3) QKV projection (WMMA, 128x128 blocks, 2x4 wave tiles; operands L2-resident)
  qkv_gemm_kernel<<<dim3((B_ * N_) / 128, QKV3_ / 128), 256, 0, stream>>>(xf, wqkvs, qf, kf, vtf);
  // 4) attention with full score panel in LDS (258 KB dynamic LDS; WGP has 320 KB)
  {
    size_t shm = (size_t)64 * 1024 * 4 + (size_t)64 * 8 * 4;
    hipFuncSetAttribute((const void*)attn_kernel, hipFuncAttributeMaxDynamicSharedMemorySize, (int)shm);
    attn_kernel<<<B_ * H_ * (N_ / 64), 256, shm, stream>>>(qf, kf, vtf, adj, dnv, aof);
  }
  // 5) output projection + bias
  proj_gemm_kernel<<<dim3((B_ * N_) / 128, DIM_ / 128), 256, 0, stream>>>(aof, wouts, b_out, out);
}
